// NeuralMemory_47493748359435
// MI455X (gfx1250) — compile-verified, matrix operand order checked by reading, not run on hardware
//
#include <hip/hip_runtime.h>

// ---------------------------------------------------------------------------
// Fused NeuralMemory forward for MI455X (gfx1250, wave32, WMMA).
//   y = ( MLP4( LN(x@wq + bq) ) ) @ w_out + b_out      (ST term == 0 in fwd)
//
// v2: BLOCK_ROWS 32 -> 64 (halves L2 weight-fragment traffic, the dominant
// cost: 256 WG x 24 MB = 6 GB from L2 vs 12 GB before). The f32 LayerNorm
// staging buffer is eliminated by computing LN statistics directly from the
// GEMM1 accumulators via ds_add_f32 atomics, freeing the LDS needed for the
// bigger activation tiles (2 x 132 KB hi/lo bf16 <= 320 KB WGP LDS).
// Precision: bf16x3 split (Ahi*Bhi + Ahi*Blo + Alo*Bhi, f32 accumulate).
// ---------------------------------------------------------------------------

#define D          1024
#define BLOCK_ROWS 64
#define WAVES      16
#define THREADS    (WAVES * 32)
#define NTW        4            // 16-wide N tiles per wave (16*4*16 = 1024 cols)
#define RT         4            // 16-high M tiles per wave  (4*16  = 64 rows)
#define KSTEPS     (D / 32)     // 32 K-steps of 32 (bf16 WMMA K)
#define NT_TOTAL   (D / 16)     // 64 N tiles
#define LDSP       1032         // padded LDS row stride (elements) - bank spread

#define SIN_BYTES  (BLOCK_ROWS * LDSP * 2)           // 132096 per hi/lo buffer
#define SRED_OFF   (2 * SIN_BYTES)                   // 264192
#define SMEM_BYTES (SRED_OFF + BLOCK_ROWS * 2 * 4)   // 264704 (<= 320KB WGP LDS)

typedef __bf16 bf16_t;
typedef bf16_t v16bf  __attribute__((ext_vector_type(16)));
typedef bf16_t bf16x8 __attribute__((ext_vector_type(8)));
typedef float  v8f    __attribute__((ext_vector_type(8)));

union V16 { v16bf v; bf16x8 h[2]; };

__device__ __forceinline__ void splitf(float x, bf16_t& hi, bf16_t& lo) {
    bf16_t h = (bf16_t)x;          // RNE to bf16
    hi = h;
    lo = (bf16_t)(x - (float)h);   // residual: next ~8 mantissa bits
}

// ---------------------------------------------------------------------------
// Prep: repack (D x D) fp32 weights (row = k, col = n) into WMMA B-fragment
// order, split into bf16 hi/lo planes.
// Fragment element index:  e = ((kt*64 + nt)*512) + lane*16 + j
// Lane l holds column n = nt*16 + (l&15); its 16 halves cover two 8-long K
// runs matching the 16-bit WMMA operand layout:
//   j<8 : k = kt*32 + (l<16 ? 0 : 8)      + j
//   j>=8: k = kt*32 + 16 + (l<16 ? 0 : 8) + (j-8)
// ---------------------------------------------------------------------------
__global__ void __launch_bounds__(256)
nm_prep_weights(const float* __restrict__ wq,
                const float* __restrict__ mlp_w,
                const float* __restrict__ w_out,
                bf16_t* __restrict__ dst)   // [6][2][D*D] bf16
{
    size_t idx = (size_t)blockIdx.x * blockDim.x + threadIdx.x;
    const size_t total = 6ull * D * D;
    if (idx >= total) return;

    const int m = (int)(idx / (D * D));    // 0=wq, 1..4=mlp, 5=w_out
    const int e = (int)(idx % (D * D));
    const float* src = (m == 0) ? wq
                     : (m <= 4) ? (mlp_w + (size_t)(m - 1) * D * D)
                                : w_out;

    const int frag = e >> 9;               // /512
    const int lane = (e >> 4) & 31;
    const int j    = e & 15;
    const int kt   = frag >> 6;            // frag = kt*64 + nt
    const int nt   = frag & 63;
    const int n    = nt * 16 + (lane & 15);
    const int kin  = (j < 8) ? ((lane < 16 ? 0 : 8) + j)
                             : (16 + (lane < 16 ? 0 : 8) + (j - 8));
    const int k    = kt * 32 + kin;

    const float v = src[(size_t)k * D + n];
    bf16_t hi, lo;
    splitf(v, hi, lo);
    dst[(size_t)m * 2 * D * D + e]         = hi;
    dst[(size_t)m * 2 * D * D + D * D + e] = lo;
}

// ---------------------------------------------------------------------------
// One 64x1024 @ 1024x1024 GEMM: A (hi/lo bf16) from LDS, B fragments (hi/lo
// bf16) from global (L2-hot), f32 accumulate via WMMA (bf16x3 scheme).
// ---------------------------------------------------------------------------
__device__ __forceinline__ void gemm_block(
    const bf16_t* __restrict__ sInHi, const bf16_t* __restrict__ sInLo,
    const bf16_t* __restrict__ wfHi,  const bf16_t* __restrict__ wfLo,
    int wave, int lane, v8f (&acc)[RT][NTW])
{
    const v8f vzero = {0.f, 0.f, 0.f, 0.f, 0.f, 0.f, 0.f, 0.f};
#pragma unroll
    for (int rt = 0; rt < RT; ++rt)
#pragma unroll
        for (int c = 0; c < NTW; ++c)
            acc[rt][c] = vzero;

    const int m    = lane & 15;
    const int koff = (lane < 16) ? 0 : 8;
    const int nt0  = wave * NTW;

#pragma unroll 1
    for (int kt = 0; kt < KSTEPS; ++kt) {
        // A fragments for the four 16-row tiles (hi & lo planes) out of LDS
        V16 aHi[RT], aLo[RT];
        const int kb = kt * 32 + koff;
#pragma unroll
        for (int rt = 0; rt < RT; ++rt) {
            const int row = rt * 16 + m;
            aHi[rt].h[0] = *(const bf16x8*)(sInHi + row * LDSP + kb);
            aHi[rt].h[1] = *(const bf16x8*)(sInHi + row * LDSP + kb + 16);
            aLo[rt].h[0] = *(const bf16x8*)(sInLo + row * LDSP + kb);
            aLo[rt].h[1] = *(const bf16x8*)(sInLo + row * LDSP + kb + 16);
        }
        // prefetch next K-step's first B fragment (global_prefetch path)
        if (kt + 1 < KSTEPS) {
            const size_t pf = (((size_t)(kt + 1) * NT_TOTAL + nt0) << 9) + lane * 16;
            __builtin_prefetch(wfHi + pf, 0, 0);
        }
#pragma unroll
        for (int c = 0; c < NTW; ++c) {
            const size_t fb = (((size_t)kt * NT_TOTAL + (nt0 + c)) << 9) + lane * 16;
            const v16bf bHi = *(const v16bf*)(wfHi + fb);
            const v16bf bLo = *(const v16bf*)(wfLo + fb);
#pragma unroll
            for (int rt = 0; rt < RT; ++rt) {
                acc[rt][c] = __builtin_amdgcn_wmma_f32_16x16x32_bf16(
                    false, aHi[rt].v, false, bHi, (short)0, acc[rt][c], false, false);
                acc[rt][c] = __builtin_amdgcn_wmma_f32_16x16x32_bf16(
                    false, aHi[rt].v, false, bLo, (short)0, acc[rt][c], false, false);
                acc[rt][c] = __builtin_amdgcn_wmma_f32_16x16x32_bf16(
                    false, aLo[rt].v, false, bHi, (short)0, acc[rt][c], false, false);
            }
        }
    }
}

// ---------------------------------------------------------------------------
// Fused forward kernel: 1 workgroup = 64 rows through all 6 GEMMs + LN + SiLU.
// ---------------------------------------------------------------------------
__global__ void __launch_bounds__(THREADS, 1)
nm_fused(const float* __restrict__ x,
         const float* __restrict__ bq,
         const float* __restrict__ mlp_b,
         const float* __restrict__ b_out,
         const bf16_t* __restrict__ wfrag,   // [6][2][D*D]
         float* __restrict__ out)
{
    extern __shared__ char smem[];
    bf16_t* sInHi = (bf16_t*)(smem);
    bf16_t* sInLo = (bf16_t*)(smem + SIN_BYTES);
    float*  sRed  = (float*)(smem + SRED_OFF);   // [64][2] sum / sumsq

    const int    tid  = threadIdx.x;
    const int    wave = tid >> 5;
    const int    lane = tid & 31;
    const int    m    = lane & 15;
    const size_t row0 = (size_t)blockIdx.x * BLOCK_ROWS;

    v8f acc[RT][NTW];

    // ---- stage x block into LDS as bf16 hi/lo; zero the LN scratch -------
    if (tid < BLOCK_ROWS * 2) sRed[tid] = 0.f;
    {
        const float4* xs = (const float4*)(x + row0 * D);
        for (int i = tid; i < BLOCK_ROWS * D / 4; i += THREADS) {
            const float4 v = xs[i];
            const int b   = i * 4;
            const int row = b >> 10;          // /D
            const int col = b & (D - 1);
            bf16_t* hp = sInHi + row * LDSP + col;
            bf16_t* lp = sInLo + row * LDSP + col;
            splitf(v.x, hp[0], lp[0]);
            splitf(v.y, hp[1], lp[1]);
            splitf(v.z, hp[2], lp[2]);
            splitf(v.w, hp[3], lp[3]);
        }
    }
    __syncthreads();

    // ---- GEMM 1: q = x @ wq + bq ; LN stats straight from accumulators ---
    gemm_block(sInHi, sInLo, wfrag, wfrag + (size_t)D * D, wave, lane, acc);
#pragma unroll
    for (int rt = 0; rt < RT; ++rt)
#pragma unroll
        for (int r = 0; r < 8; ++r) {
            const int row = rt * 16 + ((lane < 16) ? r : r + 8);
            float s = 0.f, s2 = 0.f;
#pragma unroll
            for (int c = 0; c < NTW; ++c) {
                const int col = (wave * NTW + c) * 16 + m;
                const float v = acc[rt][c][r] + bq[col];
                acc[rt][c][r] = v;            // keep biased q in registers
                s += v; s2 += v * v;
            }
            atomicAdd(&sRed[row * 2],     s);   // ds_add_f32
            atomicAdd(&sRed[row * 2 + 1], s2);
        }
    __syncthreads();   // stats complete AND all GEMM1 reads of sIn finished

    // ---- normalize in registers, split-write into sIn (hi/lo) ------------
#pragma unroll
    for (int rt = 0; rt < RT; ++rt)
#pragma unroll
        for (int r = 0; r < 8; ++r) {
            const int row  = rt * 16 + ((lane < 16) ? r : r + 8);
            const float mean = sRed[row * 2] * (1.f / D);
            const float var  = sRed[row * 2 + 1] * (1.f / D) - mean * mean;
            const float rstd = rsqrtf(var + 1e-5f);
#pragma unroll
            for (int c = 0; c < NTW; ++c) {
                const int col = (wave * NTW + c) * 16 + m;
                const float v = (acc[rt][c][r] - mean) * rstd;
                splitf(v, sInHi[row * LDSP + col], sInLo[row * LDSP + col]);
            }
        }
    __syncthreads();

    // ---- memory MLP: 4 layers, SiLU on first 3 ---------------------------
    for (int L = 0; L < 4; ++L) {
        const bf16_t* wf = wfrag + (size_t)(1 + L) * 2 * D * D;
        gemm_block(sInHi, sInLo, wf, wf + (size_t)D * D, wave, lane, acc);
        __syncthreads();   // everyone done reading sIn before overwrite
#pragma unroll
        for (int c = 0; c < NTW; ++c) {
            const int   col = (wave * NTW + c) * 16 + m;
            const float bv  = mlp_b[L * D + col];
#pragma unroll
            for (int rt = 0; rt < RT; ++rt)
#pragma unroll
                for (int r = 0; r < 8; ++r) {
                    const int row = rt * 16 + ((lane < 16) ? r : r + 8);
                    float v = acc[rt][c][r] + bv;
                    if (L < 3) v = v / (1.f + __expf(-v));   // SiLU
                    splitf(v, sInHi[row * LDSP + col], sInLo[row * LDSP + col]);
                }
        }
        __syncthreads();
    }

    // ---- final: y = h @ w_out + b_out  (ST term is 0 in forward) ---------
    gemm_block(sInHi, sInLo,
               wfrag + (size_t)5 * 2 * D * D,
               wfrag + (size_t)5 * 2 * D * D + (size_t)D * D,
               wave, lane, acc);
#pragma unroll
    for (int c = 0; c < NTW; ++c) {
        const int   col = (wave * NTW + c) * 16 + m;
        const float bv  = b_out[col];
#pragma unroll
        for (int rt = 0; rt < RT; ++rt)
#pragma unroll
            for (int r = 0; r < 8; ++r) {
                const int row = rt * 16 + ((lane < 16) ? r : r + 8);
                out[(row0 + row) * D + col] = acc[rt][c][r] + bv;
            }
    }
}

// ---------------------------------------------------------------------------
extern "C" void kernel_launch(void* const* d_in, const int* in_sizes, int n_in,
                              void* d_out, int out_size, void* d_ws, size_t ws_size,
                              hipStream_t stream) {
    (void)n_in; (void)out_size; (void)ws_size;
    const float* x     = (const float*)d_in[0];
    const float* wq    = (const float*)d_in[1];
    const float* bq    = (const float*)d_in[2];
    const float* mlp_w = (const float*)d_in[3];
    const float* mlp_b = (const float*)d_in[4];
    const float* w_out = (const float*)d_in[5];
    const float* b_out = (const float*)d_in[6];
    float*  out   = (float*)d_out;
    bf16_t* wfrag = (bf16_t*)d_ws;        // 6 * 2 * D*D bf16 = 24 MB scratch

    {   // repack weights into WMMA fragment layout (bf16 hi/lo) every call
        const size_t total = 6ull * D * D;
        dim3 grid((unsigned)((total + 255) / 256)), block(256);
        nm_prep_weights<<<grid, block, 0, stream>>>(wq, mlp_w, w_out, wfrag);
    }
    {   // fused 6-GEMM chain, 64 rows per workgroup
        const int rows = in_sizes[0] / D;            // 4*4096 = 16384
        dim3 grid(rows / BLOCK_ROWS), block(THREADS);
        nm_fused<<<grid, block, SMEM_BYTES, stream>>>(x, bq, mlp_b, b_out, wfrag, out);
    }
}
